// LSTM_46385646796958
// MI455X (gfx1250) — matrix operand order repaired and checked by reading, not compiled, not run on hardware
//
#include <hip/hip_runtime.h>
#include <hip/hip_bf16.h>

// ---------------------------------------------------------------------------
// Stacked LSTM (8 cells, B=16384, H=IN=OUT=256) + output linear, on CDNA5 WMMA.
// bf16 WMMA (v_wmma_f32_16x16x32_bf16), fp32 accumulate, fused LSTM epilogue,
// double-buffered LDS pipeline, hardware fp32->bf16 conversion.
// ---------------------------------------------------------------------------

#define BATCH 16384
#define HDIM  256
#define NLAY  8          // 1 input cell + 7 hidden cells
#define FOURH 1024

typedef __attribute__((ext_vector_type(2)))  __bf16 v2bf;
typedef __attribute__((ext_vector_type(8)))  __bf16 v8bf;
typedef __attribute__((ext_vector_type(16))) __bf16 v16bf;
typedef __attribute__((ext_vector_type(8)))  float  v8f;

__device__ __forceinline__ v16bf cat8(v8bf lo, v8bf hi) {
    return __builtin_shufflevector(lo, hi, 0,1,2,3,4,5,6,7,8,9,10,11,12,13,14,15);
}

__device__ __forceinline__ float sigf(float x) {
    return 1.0f / (1.0f + __expf(-x));
}

// register staging for one K-slice (A: 8 floats, B: 2 cols x 16 k = 32 floats)
struct Stage {
    float4 a[2];
    float4 w0[4];
    float4 w1[4];
};

// ---------------------------------------------------------------------------
// One LSTM cell:  gates = X*Wi^T + Hext*Wh^T + (bi+bh); fused i/f/g/o epilogue.
// Block tile: 64 batch rows x 64 h-columns x all 4 gates.
// grid = (BATCH/64, HDIM/64), block = 256 threads (8 waves).
// ---------------------------------------------------------------------------
__global__ __launch_bounds__(256)
void lstm_cell_wmma(const float* __restrict__ X,      // [B, 256]  cell input
                    const float* __restrict__ Hext,   // [B, 256]  h_all[l]
                    const float* __restrict__ Cext,   // [B, 256]  c_all[l]
                    const float* __restrict__ Wi,     // [1024, 256] row-major
                    const float* __restrict__ Wh,     // [1024, 256]
                    const float* __restrict__ bi,     // [1024]
                    const float* __restrict__ bh,     // [1024]
                    float* __restrict__ Hout,         // [B, 256]
                    float* __restrict__ Cout)         // [B, 256]
{
    __shared__ __bf16 ldsA[2][64][32];     //  8 KB : batch-rows x K-slice
    __shared__ __bf16 ldsB[2][32][256];    // 32 KB : K-slice x (4 gates * 64 cols)

    const int tid   = threadIdx.x;
    const int wave  = tid >> 5;
    const int lane  = tid & 31;
    const int m0    = blockIdx.x * 64;
    const int n0    = blockIdx.y * 64;       // h-column tile base
    const int mstrip = wave & 3;             // which 16-row strip of the 64
    const int npair  = wave >> 2;            // 0/1 : which pair of 16-col n-tiles

    // loader mapping (A): 4 threads per row, 8 k each
    const int la_r  = tid >> 2;
    const int la_k0 = (tid & 3) * 8;
    // loader mapping (B): thread owns 2 adjacent gate-columns x 16 k
    const int lb_col   = 2 * (tid & 127);            // 0..254, = gate*64 + i
    const int lb_khalf = (tid >> 7) * 16;            // 0 or 16
    const int lb_j     = (lb_col >> 6) * HDIM + n0 + (lb_col & 63); // global row of W

    v8f acc[2][4] = {};   // [n-subtile][gate] fp32 accumulators

    Stage st;
    // ---- software pipeline: prologue loads tile 0 ---------------------------
    {
        const float4* pa = (const float4*)(X + (size_t)(m0 + la_r) * HDIM + la_k0);
        st.a[0] = pa[0]; st.a[1] = pa[1];
        const float4* p0 = (const float4*)(Wi + (size_t)lb_j * HDIM + lb_khalf);
        const float4* p1 = (const float4*)(Wi + (size_t)(lb_j + 1) * HDIM + lb_khalf);
#pragma unroll
        for (int q = 0; q < 4; ++q) { st.w0[q] = p0[q]; st.w1[q] = p1[q]; }
    }
    // convert + store tile 0 into buffer 0
    {
        v8bf v;
        v[0]=(__bf16)st.a[0].x; v[1]=(__bf16)st.a[0].y; v[2]=(__bf16)st.a[0].z; v[3]=(__bf16)st.a[0].w;
        v[4]=(__bf16)st.a[1].x; v[5]=(__bf16)st.a[1].y; v[6]=(__bf16)st.a[1].z; v[7]=(__bf16)st.a[1].w;
        *(v8bf*)&ldsA[0][la_r][la_k0] = v;
#pragma unroll
        for (int q = 0; q < 4; ++q) {
            float4 u = st.w0[q], w = st.w1[q];
            const int k = lb_khalf + q * 4;
            *(v2bf*)&ldsB[0][k+0][lb_col] = v2bf{(__bf16)u.x, (__bf16)w.x};
            *(v2bf*)&ldsB[0][k+1][lb_col] = v2bf{(__bf16)u.y, (__bf16)w.y};
            *(v2bf*)&ldsB[0][k+2][lb_col] = v2bf{(__bf16)u.z, (__bf16)w.z};
            *(v2bf*)&ldsB[0][k+3][lb_col] = v2bf{(__bf16)u.w, (__bf16)w.w};
        }
    }

    for (int t = 0; t < 16; ++t) {
        __syncthreads();
        const int cur = t & 1;

        // ---- issue global loads for tile t+1 (latency hidden behind WMMAs) --
        if (t < 15) {
            const int  kb    = (t + 1) * 32;
            const bool xpart = (kb < 256);
            const float* asrc = xpart ? X  : Hext;
            const float* wsrc = xpart ? Wi : Wh;
            const int    kcol = xpart ? kb : (kb - 256);
            const float4* pa = (const float4*)(asrc + (size_t)(m0 + la_r) * HDIM + kcol + la_k0);
            st.a[0] = pa[0]; st.a[1] = pa[1];
            const float4* p0 = (const float4*)(wsrc + (size_t)lb_j * HDIM + kcol + lb_khalf);
            const float4* p1 = (const float4*)(wsrc + (size_t)(lb_j + 1) * HDIM + kcol + lb_khalf);
#pragma unroll
            for (int q = 0; q < 4; ++q) { st.w0[q] = p0[q]; st.w1[q] = p1[q]; }
        }

        // ---- A fragment (16x32 bf16: lane=M row, dwords pack K pairs) -------
        const int arow = mstrip * 16 + (lane & 15);
        const int ak0  = (lane < 16) ? 0 : 8;
        v16bf afrag = cat8(*(const v8bf*)&ldsA[cur][arow][ak0],
                           *(const v8bf*)&ldsA[cur][arow][ak0 + 16]);
#pragma unroll
        for (int nn = 0; nn < 2; ++nn) {
#pragma unroll
            for (int g = 0; g < 4; ++g) {
                // B fragment (32x16 bf16: lane%16=N col, elements = K in table order)
                const int col = g * 64 + (npair * 2 + nn) * 16;
                v16bf bfrag;
#pragma unroll
                for (int s = 0; s < 16; ++s)
                    bfrag[s] = ldsB[cur][(s < 8) ? (ak0 + s) : (16 + ak0 + s - 8)][col + (lane & 15)];
                acc[nn][g] = __builtin_amdgcn_wmma_f32_16x16x32_bf16(
                    false, afrag, false, bfrag, (short)0, acc[nn][g], false, false);
            }
        }

        // ---- convert + store tile t+1 into the other buffer -----------------
        if (t < 15) {
            const int nxt = cur ^ 1;
            v8bf v;
            v[0]=(__bf16)st.a[0].x; v[1]=(__bf16)st.a[0].y; v[2]=(__bf16)st.a[0].z; v[3]=(__bf16)st.a[0].w;
            v[4]=(__bf16)st.a[1].x; v[5]=(__bf16)st.a[1].y; v[6]=(__bf16)st.a[1].z; v[7]=(__bf16)st.a[1].w;
            *(v8bf*)&ldsA[nxt][la_r][la_k0] = v;
#pragma unroll
            for (int q = 0; q < 4; ++q) {
                float4 u = st.w0[q], w = st.w1[q];
                const int k = lb_khalf + q * 4;
                *(v2bf*)&ldsB[nxt][k+0][lb_col] = v2bf{(__bf16)u.x, (__bf16)w.x};
                *(v2bf*)&ldsB[nxt][k+1][lb_col] = v2bf{(__bf16)u.y, (__bf16)w.y};
                *(v2bf*)&ldsB[nxt][k+2][lb_col] = v2bf{(__bf16)u.z, (__bf16)w.z};
                *(v2bf*)&ldsB[nxt][k+3][lb_col] = v2bf{(__bf16)u.w, (__bf16)w.w};
            }
        }
    }

    // ---- fused LSTM epilogue (C/D layout: VGPR r -> M=r (+8 hi lanes),
    //      lane%16 -> N column) ----------------------------------------------
    const int ncol = lane & 15;
    const int mhi  = (lane >> 4) << 3;
#pragma unroll
    for (int nn = 0; nn < 2; ++nn) {
        const int n = n0 + (npair * 2 + nn) * 16 + ncol;
        float bsum[4];
#pragma unroll
        for (int g = 0; g < 4; ++g)
            bsum[g] = bi[g * HDIM + n] + bh[g * HDIM + n];
#pragma unroll
        for (int r = 0; r < 8; ++r) {
            const int m = m0 + mstrip * 16 + mhi + r;
            const size_t idx = (size_t)m * HDIM + n;
            float iv = acc[nn][0][r] + bsum[0];
            float fv = acc[nn][1][r] + bsum[1];
            float gv = acc[nn][2][r] + bsum[2];
            float ov = acc[nn][3][r] + bsum[3];
            float c_old = Cext[idx];
            float cn = sigf(fv) * c_old + sigf(iv) * tanhf(gv);
            float hn = sigf(ov) * tanhf(cn);
            Cout[idx] = cn;
            Hout[idx] = hn;
        }
    }
}

// ---------------------------------------------------------------------------
// Output linear: out = H * Wout^T + bout   ([16384x256] x [256x256])
// ---------------------------------------------------------------------------
__global__ __launch_bounds__(256)
void out_linear_wmma(const float* __restrict__ Hin,    // [B, 256]
                     const float* __restrict__ W,      // [256, 256] row-major
                     const float* __restrict__ bout,   // [256]
                     float* __restrict__ Out)          // [B, 256]
{
    __shared__ __bf16 ldsA[64][32];
    __shared__ __bf16 ldsB[32][64];

    const int tid  = threadIdx.x;
    const int wave = tid >> 5;
    const int lane = tid & 31;
    const int m0   = blockIdx.x * 64;
    const int n0   = blockIdx.y * 64;
    const int mstrip = wave & 3;
    const int npair  = wave >> 2;

    v8f acc[2] = {};

    for (int kb = 0; kb < 256; kb += 32) {
        {   // A tile: one packed b128 store per thread
            const int r  = tid >> 2;
            const int k0 = (tid & 3) * 8;
            const float4* p = (const float4*)(Hin + (size_t)(m0 + r) * HDIM + kb + k0);
            float4 v0 = p[0];
            float4 v1 = p[1];
            v8bf v;
            v[0]=(__bf16)v0.x; v[1]=(__bf16)v0.y; v[2]=(__bf16)v0.z; v[3]=(__bf16)v0.w;
            v[4]=(__bf16)v1.x; v[5]=(__bf16)v1.y; v[6]=(__bf16)v1.z; v[7]=(__bf16)v1.w;
            *(v8bf*)&ldsA[r][k0] = v;
        }
        {   // B tile: 32x64 K-major; thread owns 2 adjacent cols x 4 k
            const int col = 2 * (tid & 31);
            const int k0  = (tid >> 5) * 4;
            const float4 u = *(const float4*)(W + (size_t)(n0 + col) * HDIM + kb + k0);
            const float4 w = *(const float4*)(W + (size_t)(n0 + col + 1) * HDIM + kb + k0);
            *(v2bf*)&ldsB[k0+0][col] = v2bf{(__bf16)u.x, (__bf16)w.x};
            *(v2bf*)&ldsB[k0+1][col] = v2bf{(__bf16)u.y, (__bf16)w.y};
            *(v2bf*)&ldsB[k0+2][col] = v2bf{(__bf16)u.z, (__bf16)w.z};
            *(v2bf*)&ldsB[k0+3][col] = v2bf{(__bf16)u.w, (__bf16)w.w};
        }
        __syncthreads();

        const int arow = mstrip * 16 + (lane & 15);
        const int ak0  = (lane < 16) ? 0 : 8;
        v16bf afrag = cat8(*(const v8bf*)&ldsA[arow][ak0],
                           *(const v8bf*)&ldsA[arow][ak0 + 16]);
#pragma unroll
        for (int nn = 0; nn < 2; ++nn) {
            const int col = (npair * 2 + nn) * 16;
            v16bf bfrag;
#pragma unroll
            for (int s = 0; s < 16; ++s)
                bfrag[s] = ldsB[(s < 8) ? (ak0 + s) : (16 + ak0 + s - 8)][col + (lane & 15)];
            acc[nn] = __builtin_amdgcn_wmma_f32_16x16x32_bf16(
                false, afrag, false, bfrag, (short)0, acc[nn], false, false);
        }
        __syncthreads();
    }

    const int ncol = lane & 15;
    const int mhi  = (lane >> 4) << 3;
#pragma unroll
    for (int nn = 0; nn < 2; ++nn) {
        const int n = n0 + (npair * 2 + nn) * 16 + ncol;
        const float bv = bout[n];
#pragma unroll
        for (int r = 0; r < 8; ++r) {
            const int m = m0 + mstrip * 16 + mhi + r;
            Out[(size_t)m * HDIM + n] = acc[nn][r] + bv;
        }
    }
}

// ---------------------------------------------------------------------------
extern "C" void kernel_launch(void* const* d_in, const int* in_sizes, int n_in,
                              void* d_out, int out_size, void* d_ws, size_t ws_size,
                              hipStream_t stream) {
    (void)in_sizes; (void)n_in; (void)out_size; (void)d_ws; (void)ws_size;

    const float* x     = (const float*)d_in[0];
    const float* h_all = (const float*)d_in[1];   // [8, B, H]
    const float* c_all = (const float*)d_in[2];   // [8, B, H]
    const float* Wih0  = (const float*)d_in[3];
    const float* Whh0  = (const float*)d_in[4];
    const float* bih0  = (const float*)d_in[5];
    const float* bhh0  = (const float*)d_in[6];
    const float* Wih   = (const float*)d_in[7];   // [7, 1024, 256]
    const float* Whh   = (const float*)d_in[8];
    const float* bih   = (const float*)d_in[9];   // [7, 1024]
    const float* bhh   = (const float*)d_in[10];
    const float* Wout  = (const float*)d_in[11];
    const float* bout  = (const float*)d_in[12];

    float* out = (float*)d_out;                               // [B, 256]
    float* hs  = out + (size_t)BATCH * HDIM;                  // [8, B, 256]
    float* cs  = hs  + (size_t)NLAY * BATCH * HDIM;           // [8, B, 256]

    const size_t bh_stride = (size_t)BATCH * HDIM;
    const size_t w_stride  = (size_t)FOURH * HDIM;

    dim3 grid(BATCH / 64, HDIM / 64);
    dim3 block(256);

    // input cell
    lstm_cell_wmma<<<grid, block, 0, stream>>>(
        x, h_all, c_all, Wih0, Whh0, bih0, bhh0, hs, cs);

    // 7 stacked hidden cells (layer-to-layer recurrence via stream ordering)
    for (int l = 1; l < NLAY; ++l) {
        lstm_cell_wmma<<<grid, block, 0, stream>>>(
            hs + (size_t)(l - 1) * bh_stride,
            h_all + (size_t)l * bh_stride,
            c_all + (size_t)l * bh_stride,
            Wih + (size_t)(l - 1) * w_stride,
            Whh + (size_t)(l - 1) * w_stride,
            bih + (size_t)(l - 1) * FOURH,
            bhh + (size_t)(l - 1) * FOURH,
            hs + (size_t)l * bh_stride,
            cs + (size_t)l * bh_stride);
    }

    // output linear from last h
    out_linear_wmma<<<grid, block, 0, stream>>>(
        hs + (size_t)(NLAY - 1) * bh_stride, Wout, bout, out);
}
